// AsymmetricContrastiveLoss_57578331570625
// MI455X (gfx1250) — compile-verified, matrix operand order checked by reading, not run on hardware
//
#include <hip/hip_runtime.h>
#include <math.h>

typedef __attribute__((ext_vector_type(16))) __bf16       v16bf;
typedef __attribute__((ext_vector_type(8)))  float        v8f;
typedef __attribute__((ext_vector_type(8)))  unsigned int v8u;

#define BM  128   // rows (proteins) per block tile
#define BN  128   // cols (functions) per block tile
#define KB  32    // K chunk per WMMA step
#define KP2 20    // dword (bf16-pair) row stride in LDS: 80B -> 16B aligned, bank-friendly

// Split f32 into truncated-bf16 high part + bf16 residual, packed as bf16 pairs.
// hi is exact truncation => (x - hi) is exact; total product error up to ~2^-14 relative.
__device__ __forceinline__ void split2(float x, float y, unsigned& hi, unsigned& lo) {
  unsigned ux = __float_as_uint(x), uy = __float_as_uint(y);
  unsigned hx = ux & 0xffff0000u,  hy = uy & 0xffff0000u;
  float rx = x - __uint_as_float(hx);
  float ry = y - __uint_as_float(hy);
  hi = (hx >> 16) | hy;
  lo = (__float_as_uint(rx) >> 16) | (__float_as_uint(ry) & 0xffff0000u);
}

// Build a 16x32 bf16 WMMA A/B fragment from LDS per the CDNA5 VGPR layout:
// lane r (0-15): K 0..7 in v0-3, K 16..23 in v4-7; lane r+16: K 8..15 / 24..31.
// Two ds_load_b128 per fragment.
__device__ __forceinline__ v16bf ldsFrag(const unsigned* base, int r, int half) {
  const uint4* p = reinterpret_cast<const uint4*>(base + r * KP2 + half * 4);
  uint4 q0 = p[0];   // K pairs {base..base+3}
  uint4 q1 = p[2];   // +8 dwords: K pairs {base+8..base+11}
  v8u u;
  u[0] = q0.x; u[1] = q0.y; u[2] = q0.z; u[3] = q0.w;
  u[4] = q1.x; u[5] = q1.y; u[6] = q1.z; u[7] = q1.w;
  return __builtin_bit_cast(v16bf, u);
}

__global__ __launch_bounds__(256)
void contrastive_gemm_stats(const float* __restrict__ P,
                            const float* __restrict__ F,
                            const int*   __restrict__ mask,
                            const float* __restrict__ temp,
                            float4*      __restrict__ stats,
                            int B, int N, int D)
{
  // Double-buffered split-bf16 staging (2 x 40KB) + reduction scratch.
  __shared__ __attribute__((aligned(16))) unsigned aHi[2][BM * KP2];
  __shared__ __attribute__((aligned(16))) unsigned aLo[2][BM * KP2];
  __shared__ __attribute__((aligned(16))) unsigned bHi[2][BN * KP2];
  __shared__ __attribute__((aligned(16))) unsigned bLo[2][BN * KP2];
  __shared__ float red[2][BM][4];   // per wave-column partial row stats

  const int nt = blockIdx.x, mt = blockIdx.y, NT = gridDim.x;
  const int mBase = mt * BM, nBase = nt * BN;
  const int tid  = threadIdx.x;
  const int lane = tid & 31, wave = tid >> 5;     // 8 waves (wave32)
  const int wr = wave >> 1, wc = wave & 1;        // 4x2 wave grid
  const int half = lane >> 4, nIdx = lane & 15;   // WMMA lane halves

  v8f acc[2][4];
  const v8f vzero = {0.f, 0.f, 0.f, 0.f, 0.f, 0.f, 0.f, 0.f};
  #pragma unroll
  for (int i = 0; i < 2; ++i)
    #pragma unroll
    for (int j = 0; j < 4; ++j) acc[i][j] = vzero;

  // Per-thread staging registers for one K-chunk of A and B (global prefetch).
  float4 stA[4], stB[4];

  auto loadStage = [&](int kc) {
    #pragma unroll
    for (int i = 0; i < 4; ++i) {
      int idx = i * 256 + tid;
      int row = idx >> 3, q = idx & 7;   // 8 lanes cover one row's 128B chunk
      stA[i] = *reinterpret_cast<const float4*>(
          P + (size_t)(mBase + row) * D + kc + q * 4);
      stB[i] = *reinterpret_cast<const float4*>(
          F + (size_t)(nBase + row) * D + kc + q * 4);
    }
  };

  auto storeStage = [&](int buf) {
    #pragma unroll
    for (int i = 0; i < 4; ++i) {
      int idx = i * 256 + tid;
      int row = idx >> 3, q = idx & 7;
      unsigned h01, l01, h23, l23;
      split2(stA[i].x, stA[i].y, h01, l01);
      split2(stA[i].z, stA[i].w, h23, l23);
      *reinterpret_cast<uint2*>(&aHi[buf][row * KP2 + q * 2]) = make_uint2(h01, h23);
      *reinterpret_cast<uint2*>(&aLo[buf][row * KP2 + q * 2]) = make_uint2(l01, l23);
      split2(stB[i].x, stB[i].y, h01, l01);
      split2(stB[i].z, stB[i].w, h23, l23);
      *reinterpret_cast<uint2*>(&bHi[buf][row * KP2 + q * 2]) = make_uint2(h01, h23);
      *reinterpret_cast<uint2*>(&bLo[buf][row * KP2 + q * 2]) = make_uint2(l01, l23);
    }
  };

  auto compute = [&](int buf) {
    v16bf ah[2], al[2], bh[4], bl[4];
    #pragma unroll
    for (int im = 0; im < 2; ++im) {
      int r = wr * 32 + im * 16 + nIdx;
      ah[im] = ldsFrag(aHi[buf], r, half);
      al[im] = ldsFrag(aLo[buf], r, half);
    }
    #pragma unroll
    for (int in = 0; in < 4; ++in) {
      int c = wc * 64 + in * 16 + nIdx;
      bh[in] = ldsFrag(bHi[buf], c, half);
      bl[in] = ldsFrag(bLo[buf], c, half);
    }
    // 3xBF16 split GEMM: hi*hi + hi*lo + lo*hi, f32 accumulate.
    #pragma unroll
    for (int im = 0; im < 2; ++im)
      #pragma unroll
      for (int in = 0; in < 4; ++in) {
        acc[im][in] = __builtin_amdgcn_wmma_f32_16x16x32_bf16(
            false, ah[im], false, bh[in], (short)0, acc[im][in], false, false);
        acc[im][in] = __builtin_amdgcn_wmma_f32_16x16x32_bf16(
            false, ah[im], false, bl[in], (short)0, acc[im][in], false, false);
        acc[im][in] = __builtin_amdgcn_wmma_f32_16x16x32_bf16(
            false, al[im], false, bh[in], (short)0, acc[im][in], false, false);
      }
  };

  // Software-pipelined K loop: one barrier per iteration; global loads for
  // chunk kt are in flight while the WMMAs of chunk kt-1 execute, and the
  // f32->bf16 split VALU work co-executes under the matrix pipe.
  const int KT = D / KB;
  loadStage(0);
  storeStage(0);
  for (int kt = 1; kt < KT; ++kt) {
    loadStage(kt * KB);        // prefetch next chunk (global, no LDS)
    __syncthreads();           // make buffer (kt-1)&1 visible / WAR-safe
    compute((kt - 1) & 1);
    storeStage(kt & 1);        // fill the other buffer
  }
  __syncthreads();
  compute((KT - 1) & 1);

  const float invT = __expf(-temp[0]);

  // Epilogue: per-row (over this tile's 128 columns) max / sumexp / masked sum / count.
  // C-layout: lane holds col n=lane&15, rows v + 8*(lane>>4); 16-lane xor shuffles
  // reduce across the columns of one row without leaving the half-group.
  #pragma unroll
  for (int im = 0; im < 2; ++im) {
    #pragma unroll
    for (int v = 0; v < 8; ++v) {
      const int gRow = mBase + wr * 32 + im * 16 + 8 * half + v;
      float xs[4]; int mk[4];
      #pragma unroll
      for (int in = 0; in < 4; ++in) {
        xs[in] = acc[im][in][v] * invT;
        const int gCol = nBase + wc * 64 + in * 16 + nIdx;
        mk[in] = __builtin_nontemporal_load(&mask[(size_t)gRow * N + gCol]);
      }
      float mx = fmaxf(fmaxf(xs[0], xs[1]), fmaxf(xs[2], xs[3]));
      #pragma unroll
      for (int o = 1; o < 16; o <<= 1) mx = fmaxf(mx, __shfl_xor(mx, o, 32));
      float s = 0.f, ms = 0.f, cnt = 0.f;
      #pragma unroll
      for (int in = 0; in < 4; ++in) {
        s += __expf(xs[in] - mx);
        if (mk[in] == 1) { ms += xs[in]; cnt += 1.f; }
      }
      #pragma unroll
      for (int o = 1; o < 16; o <<= 1) {
        s   += __shfl_xor(s, o, 32);
        ms  += __shfl_xor(ms, o, 32);
        cnt += __shfl_xor(cnt, o, 32);
      }
      if (nIdx == 0) {
        const int rit = wr * 32 + im * 16 + 8 * half + v;
        red[wc][rit][0] = mx; red[wc][rit][1] = s;
        red[wc][rit][2] = ms; red[wc][rit][3] = cnt;
      }
    }
  }
  __syncthreads();
  // Merge the two wave-columns and emit per-(row, N-tile) stats.
  if (tid < BM) {
    float m0 = red[0][tid][0], s0 = red[0][tid][1], a0 = red[0][tid][2], c0 = red[0][tid][3];
    float m1 = red[1][tid][0], s1 = red[1][tid][1], a1 = red[1][tid][2], c1 = red[1][tid][3];
    float m = fmaxf(m0, m1);
    float s = s0 * __expf(m0 - m) + s1 * __expf(m1 - m);
    stats[(size_t)(mBase + tid) * NT + nt] = make_float4(m, s, a0 + a1, c0 + c1);
  }
}

__global__ __launch_bounds__(256)
void contrastive_row_reduce(const float4* __restrict__ stats,
                            float* __restrict__ rowLoss,
                            float* __restrict__ rowCnt,
                            int B, int NT)
{
  const int wave = threadIdx.x >> 5, lane = threadIdx.x & 31;
  const int row = blockIdx.x * 8 + wave;
  if (row >= B) return;
  float m = -INFINITY, s = 0.f, ms = 0.f, c = 0.f;
  for (int t = lane; t < NT; t += 32) {
    float4 st = stats[(size_t)row * NT + t];
    float nm = fmaxf(m, st.x);
    s = s * __expf(m - nm) + st.y * __expf(st.x - nm);
    m = nm; ms += st.z; c += st.w;
  }
  #pragma unroll
  for (int o = 16; o; o >>= 1) {
    float om = __shfl_xor(m, o, 32), os = __shfl_xor(s, o, 32);
    float nm = fmaxf(m, om);
    s = s * __expf(m - nm) + os * __expf(om - nm);
    m = nm;
    ms += __shfl_xor(ms, o, 32);
    c  += __shfl_xor(c, o, 32);
  }
  if (lane == 0) {
    float lse = m + __logf(s);
    rowLoss[row] = c * lse - ms;   // Sum over positives in this row of (lse - logit)
    rowCnt[row]  = c;
  }
}

__global__ __launch_bounds__(256)
void contrastive_final(const float* __restrict__ rowLoss,
                       const float* __restrict__ rowCnt,
                       float* __restrict__ out, int B)
{
  __shared__ float shA[256], shB[256];
  float a = 0.f, b = 0.f;
  for (int i = threadIdx.x; i < B; i += 256) { a += rowLoss[i]; b += rowCnt[i]; }
  shA[threadIdx.x] = a; shB[threadIdx.x] = b;
  __syncthreads();
  for (int ofs = 128; ofs; ofs >>= 1) {
    if (threadIdx.x < ofs) {
      shA[threadIdx.x] += shA[threadIdx.x + ofs];
      shB[threadIdx.x] += shB[threadIdx.x + ofs];
    }
    __syncthreads();
  }
  if (threadIdx.x == 0) out[0] = (shB[0] > 0.f) ? shA[0] / shB[0] : 0.f;
}

extern "C" void kernel_launch(void* const* d_in, const int* in_sizes, int n_in,
                              void* d_out, int out_size, void* d_ws, size_t ws_size,
                              hipStream_t stream) {
  const float* P    = (const float*)d_in[0];
  const float* F    = (const float*)d_in[1];
  const int*   mask = (const int*)  d_in[2];
  const float* temp = (const float*)d_in[3];

  // pb=B*D, fn=N*D, mk=B*N  ->  D = sqrt(pb*fn/mk)
  const double pb = (double)in_sizes[0], fn = (double)in_sizes[1], mk = (double)in_sizes[2];
  const int D = (int)llround(sqrt(pb * fn / mk));
  const int B = (int)llround(pb / D);
  const int N = (int)llround(fn / D);
  const int NT = N / BN;

  float4* stats   = (float4*)d_ws;                                    // [B][NT]
  float*  rowLoss = (float*)((char*)d_ws + (size_t)B * NT * sizeof(float4));
  float*  rowCnt  = rowLoss + B;

  dim3 g1(NT, B / BM);
  contrastive_gemm_stats<<<g1, 256, 0, stream>>>(P, F, mask, temp, stats, B, N, D);
  contrastive_row_reduce<<<(B + 7) / 8, 256, 0, stream>>>(stats, rowLoss, rowCnt, B, NT);
  contrastive_final<<<1, 256, 0, stream>>>(rowLoss, rowCnt, (float*)d_out, B);
}
